// A3TGCN2_EdgeClassifier_86294482911652
// MI455X (gfx1250) — matrix operand order repaired and checked by reading, not compiled, not run on hardware
//
#include <hip/hip_runtime.h>
#include <hip/hip_bf16.h>
#include <math.h>

// ---------------------------------------------------------------------------
// A3TGCN2 edge classifier forward, CDNA5 (gfx1250, wave32, WMMA 16x16x32 f16)
// Exact simplification: H0 == 0 => R-gate dead; Z = sigmoid(cz@lzW_top+lzb),
// Ht = tanh(ch@lhW_top+lhb), h = (1-Z)*Ht; softmax over 1 period == 1.0.
// B-matrix weights pre-swizzled into WMMA fragment layout in global memory;
// edge-MLP input rows gathered via async global->LDS (ASYNCcnt).
// ---------------------------------------------------------------------------

typedef __attribute__((ext_vector_type(16))) _Float16 v16h;
typedef __attribute__((ext_vector_type(8)))  _Float16 v8h;
typedef __attribute__((ext_vector_type(8)))  float    v8f;
typedef __attribute__((ext_vector_type(4)))  int      v4i;

#define AS_GLOBAL __attribute__((address_space(1)))
#define AS_LOCAL  __attribute__((address_space(3)))

#if defined(__has_builtin)
#if __has_builtin(__builtin_amdgcn_global_load_async_to_lds_b128) && \
    __has_builtin(__builtin_amdgcn_s_wait_asynccnt)
#define ASYNC_LDS_OK 1
#endif
#endif
#ifndef ASYNC_LDS_OK
#define ASYNC_LDS_OK 0
#endif

// ---- wave32 WMMA fragment helpers (layouts per CDNA5 ISA) -------

// A: 16x32 f16 tile, row-major, lda multiple of 8 halfs. Per lane this is two
// contiguous 16B chunks -> 2x ds_load_b128.
__device__ __forceinline__ v16h frag_a(const _Float16* A, int lda, int lane) {
  const int m = lane & 15;
  const int koff = (lane >> 4) << 3;
  const _Float16* row = A + m * lda + koff;
  v8h lo = *(const v8h*)row;         // K = koff .. koff+7
  v8h hi = *(const v8h*)(row + 16);  // K = 16+koff .. 16+koff+7
  return __builtin_shufflevector(lo, hi, 0, 1, 2, 3, 4, 5, 6, 7, 8, 9, 10, 11,
                                 12, 13, 14, 15);
}

// B: pre-swizzled global layout: ((kt*4+ct)*32 + lane)*16 + i, where
// element (k,n): lane = ((k>>3)&1)*16 + (n&15), i = (k&7)|(((k&31)>>4)<<3), kt = k>>5, ct=n>>4.
// One aligned 32B v16h load per lane (wave reads 1KB contiguous).
__device__ __forceinline__ v16h frag_b_swz(const _Float16* __restrict__ Wswz,
                                           int kt, int ct, int lane) {
  return *(const v16h*)(Wswz + ((((kt << 2) + ct) << 5) + lane) * 16);
}

__device__ __forceinline__ v8f wmma16x16x32(v16h a, v16h b, v8f c) {
  return __builtin_amdgcn_wmma_f32_16x16x32_f16(false, a, false, b, (short)0, c,
                                                false, false);
}

// ---------------------------------------------------------------------------
// K0: convert f32 weight (rows x 64) -> f16, zero-pad rows, swizzle into
// B-fragment layout.
__global__ void cvt_swz_b(const float* __restrict__ src,
                          _Float16* __restrict__ dst, int rows_src,
                          int rows_dst) {
  int idx = blockIdx.x * blockDim.x + threadIdx.x;
  if (idx >= rows_dst * 64) return;
  int r = idx >> 6, c = idx & 63;
  float v = (r < rows_src) ? src[r * 64 + c] : 0.0f;
  int kt = r >> 5, kin = r & 31;
  int hi = (kin >> 3) & 1, i = (kin & 7) | ((kin >> 4) << 3);
  int ct = c >> 4, n = c & 15;
  int lane = hi * 16 + n;
  dst[((((kt << 2) + ct) << 5) + lane) * 16 + i] = (_Float16)v;
}

// K1..K3: gcn_norm degree / dinv --------------------------------------------
__global__ void deg_init_kernel(float* __restrict__ deg, int N) {
  int i = blockIdx.x * blockDim.x + threadIdx.x;
  if (i < N) deg[i] = 1.0f;  // self-loop weight
}

__global__ void deg_accum_kernel(const int* __restrict__ eidx,
                                 const float* __restrict__ ew,
                                 float* __restrict__ deg, int E) {
  int e = blockIdx.x * blockDim.x + threadIdx.x;
  if (e < E) atomicAdd(&deg[eidx[E + e]], ew[e]);  // dst = edge_index[1]
}

__global__ void dinv_kernel(float* __restrict__ deg, int N) {
  int i = blockIdx.x * blockDim.x + threadIdx.x;
  if (i < N) {
    float d = deg[i];
    deg[i] = (d > 0.0f) ? rsqrtf(d) : 0.0f;  // in place: deg -> dinv
  }
}

// K4: XWz = X@Wz, XWh = X@Wh  (N x 80 @ 80 x 64, K padded to 96) -------------
__global__ void __launch_bounds__(128) node_gemm_kernel(
    const float* __restrict__ X, const _Float16* __restrict__ WzS,
    const _Float16* __restrict__ WhS, float* __restrict__ XWz,
    float* __restrict__ XWh) {
  __shared__ _Float16 Xs[16 * 96];
  const int tid = threadIdx.x;
  const int node0 = blockIdx.x * 16;
  for (int idx = tid; idx < 16 * 96; idx += 128) {
    int r = idx / 96, k = idx - r * 96;
    float v = (k < 80) ? X[(size_t)(node0 + r) * 80 + k] : 0.0f;
    Xs[idx] = (_Float16)v;
  }
  __syncthreads();
  const int wave = tid >> 5, lane = tid & 31;
  const int n0 = wave * 16;
  v8f cz = {};
  v8f ch = {};
#pragma unroll
  for (int kt = 0; kt < 3; ++kt) {
    v16h a = frag_a(Xs + kt * 32, 96, lane);
    cz = wmma16x16x32(a, frag_b_swz(WzS, kt, wave, lane), cz);
    ch = wmma16x16x32(a, frag_b_swz(WhS, kt, wave, lane), ch);
  }
  const int n = lane & 15, mh = (lane >> 4) * 8;
#pragma unroll
  for (int r = 0; r < 8; ++r) {
    size_t o = (size_t)(node0 + mh + r) * 64 + n0 + n;
    XWz[o] = cz[r];
    XWh[o] = ch[r];
  }
}

// K5a: CZ/CH = bias + self-loop term ----------------------------------------
__global__ void agg_init_kernel(const float* __restrict__ XWz,
                                const float* __restrict__ XWh,
                                const float* __restrict__ dinv,
                                const float* __restrict__ bz,
                                const float* __restrict__ bh,
                                float* __restrict__ CZ, float* __restrict__ CH,
                                int N) {
  int idx = blockIdx.x * blockDim.x + threadIdx.x;
  if (idx >= N * 64) return;
  int i = idx >> 6, f = idx & 63;
  float di = dinv[i];
  float sl = di * di;  // self-loop norm
  CZ[idx] = bz[f] + XWz[idx] * sl;
  CH[idx] = bh[f] + XWh[idx] * sl;
}

// K5b: edge scatter, one wave per edge, 2 features per lane per matrix ------
__global__ void agg_edge_kernel(const int* __restrict__ eidx,
                                const float* __restrict__ ew,
                                const float* __restrict__ dinv,
                                const float* __restrict__ XWz,
                                const float* __restrict__ XWh,
                                float* __restrict__ CZ, float* __restrict__ CH,
                                int E) {
  int gt = blockIdx.x * blockDim.x + threadIdx.x;
  int e = gt >> 5, lane = gt & 31;
  if (e >= E) return;
  int s = eidx[e];
  int d = eidx[E + e];
  float w = dinv[s] * ew[e] * dinv[d];
  size_t so = (size_t)s * 64, dofs = (size_t)d * 64;
  atomicAdd(&CZ[dofs + lane], XWz[so + lane] * w);
  atomicAdd(&CZ[dofs + 32 + lane], XWz[so + 32 + lane] * w);
  atomicAdd(&CH[dofs + lane], XWh[so + lane] * w);
  atomicAdd(&CH[dofs + 32 + lane], XWh[so + 32 + lane] * w);
}

// K6: Hn(f16) = (1 - sigmoid(CZ@Lz + lzb)) * tanh(CH@Lh + lhb) ---------------
__global__ void __launch_bounds__(128) gate_kernel(
    const float* __restrict__ CZ, const float* __restrict__ CH,
    const _Float16* __restrict__ LzS, const _Float16* __restrict__ LhS,
    const float* __restrict__ lzb, const float* __restrict__ lhb,
    _Float16* __restrict__ Hnh) {
  __shared__ _Float16 CzS[16 * 64];
  __shared__ _Float16 ChS[16 * 64];
  const int tid = threadIdx.x;
  const int node0 = blockIdx.x * 16;
  for (int idx = tid; idx < 16 * 64; idx += 128) {
    CzS[idx] = (_Float16)CZ[(size_t)node0 * 64 + idx];
    ChS[idx] = (_Float16)CH[(size_t)node0 * 64 + idx];
  }
  __syncthreads();
  const int wave = tid >> 5, lane = tid & 31;
  const int n0 = wave * 16;
  const int n = lane & 15;
  float bzv = lzb[n0 + n], bhv = lhb[n0 + n];
  v8f az, ah;
#pragma unroll
  for (int r = 0; r < 8; ++r) {
    az[r] = bzv;
    ah[r] = bhv;
  }
#pragma unroll
  for (int kt = 0; kt < 2; ++kt) {
    az = wmma16x16x32(frag_a(CzS + kt * 32, 64, lane),
                      frag_b_swz(LzS, kt, wave, lane), az);
    ah = wmma16x16x32(frag_a(ChS + kt * 32, 64, lane),
                      frag_b_swz(LhS, kt, wave, lane), ah);
  }
  const int mh = (lane >> 4) * 8;
#pragma unroll
  for (int r = 0; r < 8; ++r) {
    float z = 1.0f / (1.0f + expf(-az[r]));
    float ht = tanhf(ah[r]);
    Hnh[(size_t)(node0 + mh + r) * 64 + n0 + n] = (_Float16)((1.0f - z) * ht);
  }
}

// K7: edge classifier: out = relu([h[src],h[dst]] @ W1 + b1) @ W2 + b2 -------
__global__ void __launch_bounds__(256) edge_mlp_kernel(
    const _Float16* __restrict__ Hnh, const int* __restrict__ esrc,
    const int* __restrict__ edst, const _Float16* __restrict__ W1S,
    const float* __restrict__ b1, const float* __restrict__ W2,
    const float* __restrict__ b2, float* __restrict__ out, int E) {
  __shared__ _Float16 EIs[64 * 128];  // 64 edges x 128 feats (f16), 16 KB
  __shared__ float HID[64 * 64];      // relu hidden, 16 KB
  const int tid = threadIdx.x;
  const int e0 = blockIdx.x * 64;
  // Gather 128 f16 rows (128B each) global -> LDS; pure byte copy -> gfx1250
  // async-to-LDS path (ASYNCcnt) when the builtin is available.
  for (int idx = tid; idx < 1024; idx += 256) {
    int el = idx >> 4;          // edge within tile
    int part = (idx >> 3) & 1;  // 0 = src row, 1 = dst row
    int chunk = idx & 7;        // 16B chunk within 128B row
    int node = part ? edst[e0 + el] : esrc[e0 + el];
    const _Float16* g = Hnh + (size_t)node * 64 + chunk * 8;
    _Float16* l = &EIs[el * 128 + part * 64 + chunk * 8];
#if ASYNC_LDS_OK
    __builtin_amdgcn_global_load_async_to_lds_b128((AS_GLOBAL v4i*)g,
                                                   (AS_LOCAL v4i*)l, 0, 0);
#else
    *(v8h*)l = *(const v8h*)g;
#endif
  }
#if ASYNC_LDS_OK
  __builtin_amdgcn_s_wait_asynccnt(0);
#endif
  __syncthreads();
  const int wave = tid >> 5, lane = tid & 31;
  const int n = lane & 15, mh = (lane >> 4) * 8;
#pragma unroll
  for (int t = 0; t < 2; ++t) {
    int task = wave + t * 8;  // 16 tile-tasks over 8 waves
    int rt = task >> 2, ct = task & 3;
    float bv = b1[ct * 16 + n];
    v8f acc;
#pragma unroll
    for (int r = 0; r < 8; ++r) acc[r] = bv;
#pragma unroll
    for (int kt = 0; kt < 4; ++kt) {
      acc = wmma16x16x32(frag_a(EIs + rt * 16 * 128 + kt * 32, 128, lane),
                         frag_b_swz(W1S, kt, ct, lane), acc);
    }
#pragma unroll
    for (int r = 0; r < 8; ++r) {
      float v = acc[r];
      HID[(rt * 16 + mh + r) * 64 + ct * 16 + n] = v > 0.0f ? v : 0.0f;
    }
  }
  __syncthreads();
  if (tid < 128) {  // 64 edges x 2 outputs; f32 dot over 64
    int el = tid >> 1, j = tid & 1;
    float sum = b2[j];
#pragma unroll
    for (int f = 0; f < 64; ++f) sum += HID[el * 64 + f] * W2[f * 2 + j];
    out[(size_t)(e0 + el) * 2 + j] = sum;
  }
}

// ---------------------------------------------------------------------------
extern "C" void kernel_launch(void* const* d_in, const int* in_sizes, int n_in,
                              void* d_out, int out_size, void* d_ws,
                              size_t ws_size, hipStream_t stream) {
  const float* X    = (const float*)d_in[0];   // (1,N,80,1)
  const int*   eidx = (const int*)d_in[1];     // (2,E)
  const float* ew   = (const float*)d_in[2];   // (E,)
  const int*   esrc = (const int*)d_in[3];     // (E,)
  const int*   edst = (const int*)d_in[4];     // (E,)
  // d_in[5] attention: softmax over 1 element == 1.0 -> unused
  const float* Wz  = (const float*)d_in[6];
  const float* bz  = (const float*)d_in[7];
  const float* lzW = (const float*)d_in[8];
  const float* lzb = (const float*)d_in[9];
  // d_in[10..13]: Wr/br/lr_W/lr_b dead (H0 == 0)
  const float* Wh  = (const float*)d_in[14];
  const float* bh  = (const float*)d_in[15];
  const float* lhW = (const float*)d_in[16];
  const float* lhb = (const float*)d_in[17];
  const float* W1  = (const float*)d_in[18];
  const float* b1  = (const float*)d_in[19];
  const float* W2  = (const float*)d_in[20];
  const float* b2  = (const float*)d_in[21];

  const int N = in_sizes[0] / 80;  // 100000 (divisible by 16)
  const int E = in_sizes[2];       // 1600000 (divisible by 64)
  float* out = (float*)d_out;

  // workspace carve-out (~103 MB); all f16 blocks stay 32B-aligned
  float* deg = (float*)d_ws;          // N floats; becomes dinv in place
  float* XWz = deg + N;               // N*64
  float* XWh = XWz + (size_t)N * 64;  // N*64
  float* CZ  = XWh + (size_t)N * 64;  // N*64
  float* CH  = CZ + (size_t)N * 64;   // N*64
  _Float16* Wzh = (_Float16*)(CH + (size_t)N * 64);  // 96*64 swizzled
  _Float16* Whh = Wzh + 96 * 64;                     // 96*64 swizzled
  _Float16* Lzh = Whh + 96 * 64;                     // 64*64 swizzled (lzW top)
  _Float16* Lhh = Lzh + 64 * 64;                     // 64*64 swizzled (lhW top)
  _Float16* W1h = Lhh + 64 * 64;                     // 128*64 swizzled
  _Float16* Hnh = (_Float16*)XWz;  // reuse: XWz dead after aggregation

  // weight f16 conversion + B-fragment swizzle (K-padded where needed)
  cvt_swz_b<<<(96 * 64 + 255) / 256, 256, 0, stream>>>(Wz, Wzh, 80, 96);
  cvt_swz_b<<<(96 * 64 + 255) / 256, 256, 0, stream>>>(Wh, Whh, 80, 96);
  cvt_swz_b<<<(64 * 64 + 255) / 256, 256, 0, stream>>>(lzW, Lzh, 64, 64);
  cvt_swz_b<<<(64 * 64 + 255) / 256, 256, 0, stream>>>(lhW, Lhh, 64, 64);
  cvt_swz_b<<<(128 * 64 + 255) / 256, 256, 0, stream>>>(W1, W1h, 128, 128);

  // gcn_norm
  deg_init_kernel<<<(N + 255) / 256, 256, 0, stream>>>(deg, N);
  deg_accum_kernel<<<(E + 255) / 256, 256, 0, stream>>>(eidx, ew, deg, E);
  dinv_kernel<<<(N + 255) / 256, 256, 0, stream>>>(deg, N);

  // feature GEMMs (WMMA)
  node_gemm_kernel<<<N / 16, 128, 0, stream>>>(X, Wzh, Whh, XWz, XWh);

  // gcn aggregation (L2-resident atomics)
  agg_init_kernel<<<(N * 64 + 255) / 256, 256, 0, stream>>>(XWz, XWh, deg, bz,
                                                            bh, CZ, CH, N);
  {
    long long threads = (long long)E * 32;
    agg_edge_kernel<<<(unsigned)((threads + 255) / 256), 256, 0, stream>>>(
        eidx, ew, deg, XWz, XWh, CZ, CH, E);
  }

  // GRU gates + combine (WMMA), f16 output for the edge stage
  gate_kernel<<<N / 16, 128, 0, stream>>>(CZ, CH, Lzh, Lhh, lzb, lhb, Hnh);

  // edge classifier MLP (WMMA + async gather, dominant cost)
  edge_mlp_kernel<<<E / 64, 256, 0, stream>>>(Hnh, esrc, edst, W1h, b1, W2, b2,
                                              out, E);
}
